// TransformerGroup_35648228557122
// MI455X (gfx1250) — compile-verified, matrix operand order checked by reading, not run on hardware
//
#include <hip/hip_runtime.h>

// CDNA5 / gfx1250 point-transformer attention.
// Algebraic refactor: attn = (W_k^T q)·(NC + PE), PE folded into 5 scalars g;
// out = W_v (NC @ softmax).  Two skinny GEMMs done with v_wmma_f32_16x16x32_f16,
// streaming middle phase is the memory-bound 268MB neighbor_cost read.

typedef __attribute__((ext_vector_type(16))) _Float16 v16h;
typedef __attribute__((ext_vector_type(8)))  float    v8f;

#define BDIM  4
#define CDIM  64
#define NPTS  16384
#define SDIM  16
#define QSCALE 0.125f          // DIM_IN^-0.5
#define WAVES 4                // waves per block (128 threads)

// 16-bit A/B operand K index for element i of v16h, half h (lane>>4):
//   i<8  -> K = i + 8h          (VGPRs 0..3)
//   i>=8 -> K = i + 8 + 8h      (VGPRs 4..7)
#define KOF(i, h) ((i) + 8*(h) + (((i) >= 8) ? 8 : 0))

__global__ __launch_bounds__(WAVES * 32, 1)
void pt_attn_kernel(const float* __restrict__ xyz,
                    const float* __restrict__ nxyz,
                    const float* __restrict__ point,
                    const float* __restrict__ ncost,
                    const float* __restrict__ w_k,
                    const float* __restrict__ w_v,
                    const float* __restrict__ w_pos,
                    const float* __restrict__ b_pos,
                    float* __restrict__ out)
{
    __shared__ __attribute__((aligned(16))) float lds_qk  [WAVES][16][CDIM]; // [point in tile][channel]
    __shared__ __attribute__((aligned(16))) float lds_prod[WAVES][CDIM][SDIM];
    __shared__ __attribute__((aligned(16))) float lds_y   [WAVES][16][CDIM];
    __shared__ __attribute__((aligned(16))) float lds_g   [WAVES][16][8];

    const int tid  = threadIdx.x;
    const int wave = tid >> 5;
    const int lane = tid & 31;
    const int h    = lane >> 4;   // half-wave id
    const int sj   = lane & 15;   // column (point j) / neighbor s index

    const int tile = blockIdx.x * WAVES + wave;      // 0 .. B*N/16-1
    const int b    = tile / (NPTS / 16);
    const int n0   = (tile % (NPTS / 16)) * 16;

    //----------------------------------------------------------------------
    // Phase A: QK(64x16) = W_k^T @ Q  for 16 points; fold pos-enc into g[5].
    //----------------------------------------------------------------------
    // A operand: A[m,k] = w_k[k, m], lane holds row m = 16t + sj.
    v16h aK[4][2];
#pragma unroll
    for (int t = 0; t < 4; ++t)
#pragma unroll
        for (int ch = 0; ch < 2; ++ch)
#pragma unroll
            for (int i = 0; i < 16; ++i) {
                const int kk = ch * 32 + KOF(i, h);
                aK[t][ch][i] = (_Float16)w_k[kk * CDIM + (t * 16 + sj)];
            }

    // B operand: B[k, j] = point[b, k, n0+j] * SCALE, lane holds column sj.
    v16h bq[2];
#pragma unroll
    for (int ch = 0; ch < 2; ++ch)
#pragma unroll
        for (int i = 0; i < 16; ++i) {
            const int c = ch * 32 + KOF(i, h);
            bq[ch][i] = (_Float16)(point[(b * CDIM + c) * NPTS + n0 + sj] * QSCALE);
        }

    v8f accK[4];
#pragma unroll
    for (int t = 0; t < 4; ++t) accK[t] = (v8f){0.f,0.f,0.f,0.f,0.f,0.f,0.f,0.f};
#pragma unroll
    for (int t = 0; t < 4; ++t)
#pragma unroll
        for (int ch = 0; ch < 2; ++ch)
            accK[t] = __builtin_amdgcn_wmma_f32_16x16x32_f16(
                false, aK[t][ch], false, bq[ch], (short)0, accK[t], false, false);

    // g[d] = sum_c QK[c]*w_pos[c,d]; g4 = QK·b_pos. Lane (h,sj) owns channels
    // c = r + 8h + 16t of point sj -> per-lane partials + one xor-16 combine.
    float g0 = 0.f, g1 = 0.f, g2 = 0.f, g3 = 0.f, g4 = 0.f;
#pragma unroll
    for (int t = 0; t < 4; ++t)
#pragma unroll
        for (int r = 0; r < 8; ++r) {
            const int   c = r + 8 * h + 16 * t;
            const float q = accK[t][r];
            g0 += q * w_pos[c * 4 + 0];
            g1 += q * w_pos[c * 4 + 1];
            g2 += q * w_pos[c * 4 + 2];
            g3 += q * w_pos[c * 4 + 3];
            g4 += q * b_pos[c];
            lds_qk[wave][sj][c] = q;
        }
    g0 += __shfl_xor(g0, 16); g1 += __shfl_xor(g1, 16);
    g2 += __shfl_xor(g2, 16); g3 += __shfl_xor(g3, 16);
    g4 += __shfl_xor(g4, 16);
    if (h == 0) {
        lds_g[wave][sj][0] = g0; lds_g[wave][sj][1] = g1;
        lds_g[wave][sj][2] = g2; lds_g[wave][sj][3] = g3;
        lds_g[wave][sj][4] = g4;
    }
    __builtin_amdgcn_wave_barrier();
    asm volatile("s_wait_dscnt 0" ::: "memory");   // wave-private LDS RAW fence

    //----------------------------------------------------------------------
    // Phase B: per point — scores, softmax over S, y = NC @ p  (streaming).
    //----------------------------------------------------------------------
#pragma unroll 1
    for (int j = 0; j < 16; ++j) {
        const int n = n0 + j;

        // geometry for neighbor s = sj (both halves redundantly)
        const float t0 = xyz[(b * 3 + 0) * NPTS + n] - nxyz[((b * 3 + 0) * NPTS + n) * SDIM + sj];
        const float t1 = xyz[(b * 3 + 1) * NPTS + n] - nxyz[((b * 3 + 1) * NPTS + n) * SDIM + sj];
        const float t2 = xyz[(b * 3 + 2) * NPTS + n] - nxyz[((b * 3 + 2) * NPTS + n) * SDIM + sj];
        const float nrm = sqrtf(t0 * t0 + t1 * t1 + t2 * t2);

        float qk[32];
#pragma unroll
        for (int q4 = 0; q4 < 8; ++q4) {
            const float4 v = *(const float4*)&lds_qk[wave][j][h * 32 + q4 * 4];
            qk[q4 * 4 + 0] = v.x; qk[q4 * 4 + 1] = v.y;
            qk[q4 * 4 + 2] = v.z; qk[q4 * 4 + 3] = v.w;
        }

        // coalesced: each load inst covers two full 64B channel rows
        float nc[32];
        float dot = 0.f;
        const size_t base = (((size_t)b * CDIM + h * 32) * NPTS + (size_t)n) * SDIM + sj;
#pragma unroll
        for (int cc = 0; cc < 32; ++cc) {
            nc[cc] = ncost[base + (size_t)cc * ((size_t)NPTS * SDIM)];
            dot += qk[cc] * nc[cc];
        }
        dot += __shfl_xor(dot, 16);

        const float logit = dot
            + lds_g[wave][j][0] * t0 + lds_g[wave][j][1] * t1
            + lds_g[wave][j][2] * t2 + lds_g[wave][j][3] * nrm
            + lds_g[wave][j][4];

        // softmax across the 16-lane group (xor butterflies stay inside group)
        float mx = logit;
        mx = fmaxf(mx, __shfl_xor(mx, 8));
        mx = fmaxf(mx, __shfl_xor(mx, 4));
        mx = fmaxf(mx, __shfl_xor(mx, 2));
        mx = fmaxf(mx, __shfl_xor(mx, 1));
        const float e = __expf(logit - mx);
        float sm = e;
        sm += __shfl_xor(sm, 8);
        sm += __shfl_xor(sm, 4);
        sm += __shfl_xor(sm, 2);
        sm += __shfl_xor(sm, 1);
        const float p = e / sm;

        // transpose p*NC through LDS to reduce over s
#pragma unroll
        for (int cc = 0; cc < 32; ++cc)
            lds_prod[wave][h * 32 + cc][sj] = nc[cc] * p;
        __builtin_amdgcn_wave_barrier();
        asm volatile("s_wait_dscnt 0" ::: "memory");

        float y0 = 0.f, y1 = 0.f;
#pragma unroll
        for (int s4 = 0; s4 < 4; ++s4) {
            const float4 r0 = *(const float4*)&lds_prod[wave][lane][s4 * 4];
            const float4 r1 = *(const float4*)&lds_prod[wave][lane + 32][s4 * 4];
            y0 += r0.x + r0.y + r0.z + r0.w;
            y1 += r1.x + r1.y + r1.z + r1.w;
        }
        lds_y[wave][j][lane]      = y0;
        lds_y[wave][j][lane + 32] = y1;
        __builtin_amdgcn_wave_barrier();
        asm volatile("s_wait_dscnt 0" ::: "memory");  // before next j reuses lds_prod
    }

    //----------------------------------------------------------------------
    // Phase C: OUT(64x16) = W_v @ Y  for the 16 points.
    //----------------------------------------------------------------------
    v16h aV[4][2];
#pragma unroll
    for (int t = 0; t < 4; ++t)
#pragma unroll
        for (int ch = 0; ch < 2; ++ch)
#pragma unroll
            for (int i = 0; i < 16; ++i) {
                const int kk = ch * 32 + KOF(i, h);
                aV[t][ch][i] = (_Float16)w_v[(t * 16 + sj) * CDIM + kk];
            }

    v16h by[2];
#pragma unroll
    for (int ch = 0; ch < 2; ++ch)
#pragma unroll
        for (int i = 0; i < 16; ++i) {
            const int c = ch * 32 + KOF(i, h);
            by[ch][i] = (_Float16)lds_y[wave][sj][c];
        }

    v8f accO[4];
#pragma unroll
    for (int t = 0; t < 4; ++t) accO[t] = (v8f){0.f,0.f,0.f,0.f,0.f,0.f,0.f,0.f};
#pragma unroll
    for (int t = 0; t < 4; ++t)
#pragma unroll
        for (int ch = 0; ch < 2; ++ch)
            accO[t] = __builtin_amdgcn_wmma_f32_16x16x32_f16(
                false, aV[t][ch], false, by[ch], (short)0, accO[t], false, false);

#pragma unroll
    for (int t = 0; t < 4; ++t)
#pragma unroll
        for (int r = 0; r < 8; ++r) {
            const int c = r + 8 * h + 16 * t;
            out[(b * CDIM + c) * NPTS + n0 + sj] = accO[t][r];
        }
}

extern "C" void kernel_launch(void* const* d_in, const int* in_sizes, int n_in,
                              void* d_out, int out_size, void* d_ws, size_t ws_size,
                              hipStream_t stream) {
    const float* xyz   = (const float*)d_in[0];
    const float* nxyz  = (const float*)d_in[1];
    const float* point = (const float*)d_in[2];
    // d_in[3] = neighbor_points (unused by the reference)
    const float* ncost = (const float*)d_in[4];
    const float* w_k   = (const float*)d_in[5];
    const float* w_v   = (const float*)d_in[6];
    const float* w_pos = (const float*)d_in[7];
    const float* b_pos = (const float*)d_in[8];
    float* outp = (float*)d_out;

    const int tiles  = BDIM * (NPTS / 16);   // 4096 sixteen-point tiles
    const int blocks = tiles / WAVES;        // 1024 blocks of 4 waves
    hipLaunchKernelGGL(pt_attn_kernel, dim3(blocks), dim3(WAVES * 32), 0, stream,
                       xyz, nxyz, point, ncost, w_k, w_v, w_pos, b_pos, outp);
}
